// CPC_9698036154560
// MI455X (gfx1250) — compile-verified, matrix operand order checked by reading, not run on hardware
//
#include <hip/hip_runtime.h>
#include <math.h>

// ---------------- types for CDNA5 WMMA ----------------
typedef __bf16 bf16;
typedef __attribute__((ext_vector_type(8)))  __bf16 v8bf;
typedef __attribute__((ext_vector_type(16))) __bf16 v16bf;
typedef __attribute__((ext_vector_type(8)))  float  v8f;

#define WMMA_BF16(a, b, c) \
  __builtin_amdgcn_wmma_f32_16x16x32_bf16(false, (a), false, (b), (short)0, (c), false, false)

// Fragment loader for 16-bit 16x32 operand tiles (A layout; B uses it on B^T).
// Row-major matrix P with leading dim ld (elements). Tile rows r0..r0+15, K k0..k0+31.
// lane 0-15: row = r0+lane, K-base 0; lane 16-31: row = r0+lane-16, K-base +8.
// Per lane the 16 elements are two contiguous 8-element runs -> two 128-bit loads.
__device__ __forceinline__ v16bf load_frag(const bf16* __restrict__ P, int ld,
                                           int r0, int k0, int lane) {
  const bf16* row = P + (size_t)(r0 + (lane & 15)) * ld + k0 + ((lane & 16) ? 8 : 0);
  v8bf lo = *(const v8bf*)row;         // K = kb+0 .. kb+7
  v8bf hi = *(const v8bf*)(row + 16);  // K = 16+kb+0 .. 16+kb+7
  return __builtin_shufflevector(lo, hi, 0, 1, 2, 3, 4, 5, 6, 7,
                                 8, 9, 10, 11, 12, 13, 14, 15);
}

// ---------------- generic bf16 GEMM: C = A(MxK) * Bt^T + bias, optional relu ----
// Bt is the (N,K) row-major weight (= B^T), exactly the natural OIHW flatten.
// block = 128 threads (4 waves); wave tile = 16M x 64N; M%64==0, N%64==0, K%64==0.
// Software-pipelined: next K-step fragments are loaded before current WMMAs issue.
__global__ void gemm_bf16_kernel(const bf16* __restrict__ A, const bf16* __restrict__ Bt,
                                 const float* __restrict__ bias,
                                 float* __restrict__ Cf, bf16* __restrict__ Cbf,
                                 int M, int N, int K, int do_relu) {
  const int lane = threadIdx.x & 31;
  const int wave = threadIdx.x >> 5;
  const int m0 = (blockIdx.y * 4 + wave) * 16;
  const int n0 = blockIdx.x * 64;
  if (m0 >= M) return;
  v8f acc[4] = {};
  v16bf a_cur = load_frag(A, K, m0, 0, lane);
  v16bf b_cur[4];
#pragma unroll
  for (int t = 0; t < 4; ++t) b_cur[t] = load_frag(Bt, K, n0 + t * 16, 0, lane);
  for (int k0 = 0; k0 + 32 < K; k0 += 32) {
    const int k1 = k0 + 32;
    // near-cache prefetch of the K-step after next (gfx1250 global_prefetch_b8)
    __builtin_prefetch(A + (size_t)(m0 + (lane & 15)) * K + k1 + 32, 0, 3);
    v16bf a_nxt = load_frag(A, K, m0, k1, lane);
    v16bf b_nxt[4];
#pragma unroll
    for (int t = 0; t < 4; ++t) b_nxt[t] = load_frag(Bt, K, n0 + t * 16, k1, lane);
#pragma unroll
    for (int t = 0; t < 4; ++t) acc[t] = WMMA_BF16(a_cur, b_cur[t], acc[t]);
    a_cur = a_nxt;
#pragma unroll
    for (int t = 0; t < 4; ++t) b_cur[t] = b_nxt[t];
  }
#pragma unroll
  for (int t = 0; t < 4; ++t) acc[t] = WMMA_BF16(a_cur, b_cur[t], acc[t]);

  const int row_base = m0 + ((lane & 16) ? 8 : 0);
#pragma unroll
  for (int t = 0; t < 4; ++t) {
    const int col = n0 + t * 16 + (lane & 15);
    const float bb = bias ? bias[col] : 0.0f;
#pragma unroll
    for (int v = 0; v < 8; ++v) {
      float val = acc[t][v] + bb;
      if (do_relu) val = fmaxf(val, 0.0f);
      const size_t off = (size_t)(row_base + v) * N + col;
      if (Cf)  Cf[off]  = val;
      if (Cbf) Cbf[off] = (bf16)val;
    }
  }
}

// ---------------- fused encoder conv2 (1x1, 256->2048) + bias + relu + mean pool ----
// H: (3136*64, 256) bf16. W2t: (2048, 256) bf16. Per wave: one patch x 16 out channels.
__global__ void gemm2_relu_pool_kernel(const bf16* __restrict__ H, const bf16* __restrict__ W2t,
                                       const float* __restrict__ b2,
                                       float* __restrict__ latf, bf16* __restrict__ latb) {
  const int lane = threadIdx.x & 31;
  const int wave = threadIdx.x >> 5;
  const int tile = blockIdx.x * 4 + wave;      // 3136 * 128 tiles
  const int patch = tile >> 7;
  const int n0 = (tile & 127) * 16;
  const bf16* A = H + (size_t)patch * 64 * 256;
  v8f acc[4] = {};
  v16bf b_cur = load_frag(W2t, 256, n0, 0, lane);
  v16bf a_cur[4];
#pragma unroll
  for (int t = 0; t < 4; ++t) a_cur[t] = load_frag(A, 256, t * 16, 0, lane);
  for (int k0 = 0; k0 + 32 < 256; k0 += 32) {
    const int k1 = k0 + 32;
    v16bf b_nxt = load_frag(W2t, 256, n0, k1, lane);
    v16bf a_nxt[4];
#pragma unroll
    for (int t = 0; t < 4; ++t) a_nxt[t] = load_frag(A, 256, t * 16, k1, lane);
#pragma unroll
    for (int t = 0; t < 4; ++t) acc[t] = WMMA_BF16(a_cur[t], b_cur, acc[t]);
    b_cur = b_nxt;
#pragma unroll
    for (int t = 0; t < 4; ++t) a_cur[t] = a_nxt[t];
  }
#pragma unroll
  for (int t = 0; t < 4; ++t) acc[t] = WMMA_BF16(a_cur[t], b_cur, acc[t]);

  const int ncol = n0 + (lane & 15);
  const float bias = b2[ncol];
  float s = 0.0f;
#pragma unroll
  for (int t = 0; t < 4; ++t)
#pragma unroll
    for (int v = 0; v < 8; ++v) s += fmaxf(acc[t][v] + bias, 0.0f);
  s += __shfl_xor(s, 16);                       // combine M halves (wave32)
  if (lane < 16) {
    const float m = s * (1.0f / 64.0f);
    const size_t off = (size_t)patch * 2048 + ncol;
    latf[off] = m;
    latb[off] = (bf16)m;
  }
}

// ---------------- masked patch im2col for encoder conv1 ----------------
// A1 row m = patch*64 + oy*8+ox ; col k = ch*64 + ky*8 + kx (matches OIHW flatten)
__global__ void patch_im2col_kernel(const float* __restrict__ img, const int* __restrict__ rnd,
                                    bf16* __restrict__ A1) {
  const size_t total = (size_t)200704 * 192;
  for (size_t idx = blockIdx.x * (size_t)blockDim.x + threadIdx.x; idx < total;
       idx += (size_t)gridDim.x * blockDim.x) {
    const int m = (int)(idx / 192);
    const int k = (int)(idx % 192);
    const int ch = k >> 6, ky = (k >> 3) & 7, kx = k & 7;
    const int pos = m & 63, np = m >> 6;
    const int oy = pos >> 3, ox = pos & 7;
    const int b = np / 49, rc = np % 49;
    const int pr = rc / 7, pcc = rc % 7;
    const int y = oy * 8 + ky, x = ox * 8 + kx;
    const int rr = rnd[np] >> 2, cc = rnd[np] & 3;
    float v = -1.0f;
    if (y >= rr && y < rr + 60 && x >= cc && x < cc + 60)
      v = img[(((size_t)b * 3 + ch) * 256 + (pr * 32 + y)) * 256 + (pcc * 32 + x)];
    A1[idx] = (bf16)v;
  }
}

// ---------------- flat f32 -> bf16 convert (weights keep natural (N,K) layout) ----
__global__ void f32_to_bf16_kernel(const float* __restrict__ in, bf16* __restrict__ out,
                                   size_t n) {
  for (size_t i = blockIdx.x * (size_t)blockDim.x + threadIdx.x; i < n;
       i += (size_t)gridDim.x * blockDim.x)
    out[i] = (bf16)in[i];
}

__global__ void copy_lat_kernel(const float* __restrict__ lf, const bf16* __restrict__ lb,
                                float* __restrict__ xf, bf16* __restrict__ xb, int n) {
  for (int i = blockIdx.x * blockDim.x + threadIdx.x; i < n; i += gridDim.x * blockDim.x) {
    xf[i] = lf[i];
    xb[i] = lb[i];
  }
}

// im2col for 1x3 conv over W (pad 1 both sides): (3136,256) -> (3136, 768), k = i*3+kx
__global__ void im2col_w_kernel(const bf16* __restrict__ Y, bf16* __restrict__ Acol) {
  const int total = 3136 * 768;
  for (int idx = blockIdx.x * blockDim.x + threadIdx.x; idx < total;
       idx += gridDim.x * blockDim.x) {
    const int m = idx / 768, k = idx % 768;
    const int i = k / 3, kx = k % 3;
    const int b = m / 49, rem = m % 49;
    const int h = rem / 7, w = rem % 7;
    const int ws = w + kx - 1;
    bf16 v = (bf16)0.0f;
    if (ws >= 0 && ws < 7) v = Y[(size_t)(b * 49 + h * 7 + ws) * 256 + i];
    Acol[idx] = v;
  }
}

// im2col for 2x1 causal conv over H (pad 1 top): (3136,256) -> (3136, 512), k = i*2+ky
__global__ void im2col_h_kernel(const bf16* __restrict__ Y, bf16* __restrict__ Acol) {
  const int total = 3136 * 512;
  for (int idx = blockIdx.x * blockDim.x + threadIdx.x; idx < total;
       idx += gridDim.x * blockDim.x) {
    const int m = idx / 512, k = idx % 512;
    const int i = k >> 1, ky = k & 1;
    const int b = m / 49, rem = m % 49;
    const int h = rem / 7, w = rem % 7;
    const int hs = h + ky - 1;
    bf16 v = (bf16)0.0f;
    if (hs >= 0) v = Y[(size_t)(b * 49 + hs * 7 + w) * 256 + i];
    Acol[idx] = v;
  }
}

__global__ void residual_kernel(const float* __restrict__ y4, float* __restrict__ xf,
                                bf16* __restrict__ xb, int n) {
  for (int i = blockIdx.x * blockDim.x + threadIdx.x; i < n; i += gridDim.x * blockDim.x) {
    const float v = fmaxf(y4[i] + xf[i], 0.0f);
    xf[i] = v;
    xb[i] = (bf16)v;
  }
}

// compact valid rows (h < ci) and apply EMB_SCALE, convert to bf16
__global__ void gather_preds_kernel(const float* __restrict__ P, bf16* __restrict__ PA, int ci) {
  const int rows = ci * 7;
  const int total = 64 * rows * 64;
  for (int idx = blockIdx.x * blockDim.x + threadIdx.x; idx < total;
       idx += gridDim.x * blockDim.x) {
    const int t = idx >> 6, kk = idx & 63;
    const int b = t / rows, rem = t % rows;
    const int src = b * 49 + rem;
    PA[idx] = (bf16)(0.1f * P[(size_t)src * 64 + kk]);
  }
}

// one block (128 thr) per logits row: -log_softmax at label, mean-accumulated into loss
__global__ void loss_kernel(const float* __restrict__ logits, int total, int ci, int label_off,
                            float* __restrict__ loss) {
  __shared__ float sm[128];
  const int row = blockIdx.x;
  const int rows = ci * 7;
  const int b = row / rows, rem = row % rows;
  const int label = b * 49 + label_off + rem;
  const float* L = logits + (size_t)row * 3136;
  float mx = -3.0e38f;
  for (int j = threadIdx.x; j < 3136; j += 128) mx = fmaxf(mx, L[j]);
  sm[threadIdx.x] = mx;
  __syncthreads();
  for (int s = 64; s > 0; s >>= 1) {
    if (threadIdx.x < s) sm[threadIdx.x] = fmaxf(sm[threadIdx.x], sm[threadIdx.x + s]);
    __syncthreads();
  }
  mx = sm[0];
  __syncthreads();
  float se = 0.0f;
  for (int j = threadIdx.x; j < 3136; j += 128) se += __expf(L[j] - mx);
  sm[threadIdx.x] = se;
  __syncthreads();
  for (int s = 64; s > 0; s >>= 1) {
    if (threadIdx.x < s) sm[threadIdx.x] += sm[threadIdx.x + s];
    __syncthreads();
  }
  if (threadIdx.x == 0) {
    const float nll = mx + __logf(sm[0]) - L[label];
    atomicAdd(loss, nll / (float)total);
  }
}

__global__ void latmean_kernel(const float* __restrict__ lat, float* __restrict__ out) {
  const int total = 64 * 2048;
  for (int idx = blockIdx.x * blockDim.x + threadIdx.x; idx < total;
       idx += gridDim.x * blockDim.x) {
    const int b = idx >> 11, ch = idx & 2047;
    float s = 0.0f;
    for (int p = 0; p < 49; ++p) s += lat[(size_t)(b * 49 + p) * 2048 + ch];
    out[1 + idx] = s * (1.0f / 49.0f);
  }
}

__global__ void zero_loss_kernel(float* __restrict__ out) {
  if (threadIdx.x == 0 && blockIdx.x == 0) out[0] = 0.0f;
}

// ======================================================================
extern "C" void kernel_launch(void* const* d_in, const int* in_sizes, int n_in,
                              void* d_out, int out_size, void* d_ws, size_t ws_size,
                              hipStream_t stream) {
  const float* images = (const float*)d_in[0];
  const int*   rnd    = (const int*)d_in[1];
  const float* enc_w1 = (const float*)d_in[2];
  const float* enc_b1 = (const float*)d_in[3];
  const float* enc_w2 = (const float*)d_in[4];
  const float* enc_b2 = (const float*)d_in[5];
  const float* pc_w1  = (const float*)d_in[6];
  const float* pc_b1  = (const float*)d_in[7];
  const float* pc_w2  = (const float*)d_in[8];
  const float* pc_b2  = (const float*)d_in[9];
  const float* pc_w3  = (const float*)d_in[10];
  const float* pc_b3  = (const float*)d_in[11];
  const float* pc_w4  = (const float*)d_in[12];
  const float* pc_b4  = (const float*)d_in[13];
  const float* z2t_w  = (const float*)d_in[14];
  const float* z2t_b  = (const float*)d_in[15];
  const float* c2p_w  = (const float*)d_in[16];
  const float* c2p_b  = (const float*)d_in[17];
  float* out = (float*)d_out;

  char* w = (char*)d_ws;
  auto alloc = [&](size_t bytes) -> char* {
    char* p = w;
    w += (bytes + 255) & ~(size_t)255;
    return p;
  };
  bf16*  A1     = (bf16*)alloc((size_t)200704 * 192 * 2);
  bf16*  H1     = (bf16*)alloc((size_t)200704 * 256 * 2);
  float* latf   = (float*)alloc((size_t)3136 * 2048 * 4);
  bf16*  latb   = (bf16*)alloc((size_t)3136 * 2048 * 2);
  float* xf     = (float*)alloc((size_t)3136 * 2048 * 4);
  bf16*  xb     = (bf16*)alloc((size_t)3136 * 2048 * 2);
  bf16*  y1b    = (bf16*)alloc((size_t)3136 * 256 * 2);
  bf16*  y1col  = (bf16*)alloc((size_t)3136 * 768 * 2);
  bf16*  y2b    = (bf16*)alloc((size_t)3136 * 256 * 2);
  bf16*  y2col  = (bf16*)alloc((size_t)3136 * 512 * 2);
  bf16*  y3b    = (bf16*)alloc((size_t)3136 * 256 * 2);
  float* y4f    = (float*)alloc((size_t)3136 * 2048 * 4);
  bf16*  W1b    = (bf16*)alloc((size_t)256 * 192 * 2);        // (N,K) natural layouts
  bf16*  W2b    = (bf16*)alloc((size_t)2048 * 256 * 2);
  bf16*  PCW1b  = (bf16*)alloc((size_t)5 * 256 * 2048 * 2);
  bf16*  PCW2b  = (bf16*)alloc((size_t)5 * 256 * 768 * 2);
  bf16*  PCW3b  = (bf16*)alloc((size_t)5 * 256 * 512 * 2);
  bf16*  PCW4b  = (bf16*)alloc((size_t)5 * 2048 * 256 * 2);
  bf16*  Z2Tb   = (bf16*)alloc((size_t)64 * 2048 * 2);
  bf16*  C2Pb   = (bf16*)alloc((size_t)3 * 64 * 2048 * 2);
  float* targf  = (float*)alloc((size_t)3136 * 64 * 4);
  bf16*  targb  = (bf16*)alloc((size_t)3136 * 64 * 2);        // (N=3136, K=64) = B^T row-major
  float* predsf = (float*)alloc((size_t)3136 * 64 * 4);
  bf16*  predsA = (bf16*)alloc((size_t)1792 * 64 * 2);
  float* logits = (float*)alloc((size_t)1792 * 3136 * 4);
  (void)ws_size; (void)in_sizes; (void)n_in; (void)out_size;

  const dim3 B256(256), B128(128);
  auto ew = [](size_t n) { return dim3((unsigned)((n + 255) / 256)); };

  zero_loss_kernel<<<1, 32, 0, stream>>>(out);

  // --- weight conversions: straight f32->bf16, natural (N,K) layout kept ---
  f32_to_bf16_kernel<<<ew(256 * 192), B256, 0, stream>>>(enc_w1, W1b, (size_t)256 * 192);
  f32_to_bf16_kernel<<<ew(2048 * 256), B256, 0, stream>>>(enc_w2, W2b, (size_t)2048 * 256);
  f32_to_bf16_kernel<<<ew((size_t)5 * 256 * 2048), B256, 0, stream>>>(pc_w1, PCW1b,
                                                                      (size_t)5 * 256 * 2048);
  f32_to_bf16_kernel<<<ew((size_t)5 * 256 * 768), B256, 0, stream>>>(pc_w2, PCW2b,
                                                                     (size_t)5 * 256 * 768);
  f32_to_bf16_kernel<<<ew((size_t)5 * 256 * 512), B256, 0, stream>>>(pc_w3, PCW3b,
                                                                     (size_t)5 * 256 * 512);
  f32_to_bf16_kernel<<<ew((size_t)5 * 2048 * 256), B256, 0, stream>>>(pc_w4, PCW4b,
                                                                      (size_t)5 * 2048 * 256);
  f32_to_bf16_kernel<<<ew(64 * 2048), B256, 0, stream>>>(z2t_w, Z2Tb, (size_t)64 * 2048);
  f32_to_bf16_kernel<<<ew((size_t)3 * 64 * 2048), B256, 0, stream>>>(c2p_w, C2Pb,
                                                                     (size_t)3 * 64 * 2048);

  // --- encoder ---
  patch_im2col_kernel<<<ew((size_t)200704 * 192), B256, 0, stream>>>(images, rnd, A1);
  gemm_bf16_kernel<<<dim3(256 / 64, 200704 / 64), B128, 0, stream>>>(
      A1, W1b, enc_b1, nullptr, H1, 200704, 256, 192, 1);
  gemm2_relu_pool_kernel<<<dim3(3136 * 128 / 4), B128, 0, stream>>>(H1, W2b, enc_b2, latf, latb);
  copy_lat_kernel<<<ew(3136 * 2048), B256, 0, stream>>>(latf, latb, xf, xb, 3136 * 2048);

  // --- PixelCNN: 5 residual blocks ---
  for (int k = 0; k < 5; ++k) {
    gemm_bf16_kernel<<<dim3(256 / 64, 3136 / 64), B128, 0, stream>>>(
        xb, PCW1b + (size_t)k * 256 * 2048, pc_b1 + k * 256, nullptr, y1b, 3136, 256, 2048, 1);
    im2col_w_kernel<<<ew(3136 * 768), B256, 0, stream>>>(y1b, y1col);
    gemm_bf16_kernel<<<dim3(256 / 64, 3136 / 64), B128, 0, stream>>>(
        y1col, PCW2b + (size_t)k * 256 * 768, pc_b2 + k * 256, nullptr, y2b, 3136, 256, 768, 1);
    im2col_h_kernel<<<ew(3136 * 512), B256, 0, stream>>>(y2b, y2col);
    gemm_bf16_kernel<<<dim3(256 / 64, 3136 / 64), B128, 0, stream>>>(
        y2col, PCW3b + (size_t)k * 256 * 512, pc_b3 + k * 256, nullptr, y3b, 3136, 256, 512, 1);
    gemm_bf16_kernel<<<dim3(2048 / 64, 3136 / 64), B128, 0, stream>>>(
        y3b, PCW4b + (size_t)k * 2048 * 256, pc_b4 + k * 2048, y4f, nullptr, 3136, 2048, 256, 0);
    residual_kernel<<<ew(3136 * 2048), B256, 0, stream>>>(y4f, xf, xb, 3136 * 2048);
  }

  // --- targets ---
  gemm_bf16_kernel<<<dim3(64 / 64, 3136 / 64), B128, 0, stream>>>(
      latb, Z2Tb, z2t_b, targf, nullptr, 3136, 64, 2048, 0);
  f32_to_bf16_kernel<<<ew((size_t)3136 * 64), B256, 0, stream>>>(targf, targb,
                                                                 (size_t)3136 * 64);

  // --- contrastive loss over 3 prediction steps (i = 2,3,4) ---
  for (int s = 0; s < 3; ++s) {
    const int i = s + 2;
    const int ci = 6 - i;            // rows kept along H
    const int total = 64 * ci * 7;   // 1792, 1344, 896 — all %64==0
    gemm_bf16_kernel<<<dim3(64 / 64, 3136 / 64), B128, 0, stream>>>(
        xb, C2Pb + (size_t)s * 64 * 2048, c2p_b + s * 64, predsf, nullptr, 3136, 64, 2048, 0);
    gather_preds_kernel<<<ew((size_t)total * 64), B256, 0, stream>>>(predsf, predsA, ci);
    gemm_bf16_kernel<<<dim3(3136 / 64, total / 64), B128, 0, stream>>>(
        predsA, targb, nullptr, logits, nullptr, total, 3136, 64, 0);
    loss_kernel<<<dim3(total), B128, 0, stream>>>(logits, total, ci, (i + 1) * 7, out);
  }

  // --- second output: latents spatial mean (64, 2048) ---
  latmean_kernel<<<ew(64 * 2048), B256, 0, stream>>>(latf, out);
}
